// AttentionDecoder_68624987455999
// MI455X (gfx1250) — matrix operand order repaired and checked by reading, not compiled, not run on hardware
//
#include <hip/hip_runtime.h>
#include <hip/hip_bf16.h>
#include <math.h>

// Problem dims (fixed by setup_inputs)
#define Bd 32
#define Kd 64
#define Ed 512
#define Md 512
#define Hd 512
#define Td 20
#define Vd 32000
#define KL 2560   // K-dim of the big Wl GEMM
#define KG 1536   // E + M + H  : K-dim of the gates GEMM
#define NG 2048   // 4H         : N-dim of the gates GEMM

typedef __attribute__((ext_vector_type(16))) __bf16 bf16x16;
typedef __attribute__((ext_vector_type(8)))  __bf16 bf16x8;
typedef __attribute__((ext_vector_type(8)))  float  f32x8;

// ---------- helpers ----------
__device__ __forceinline__ unsigned short f2bf(float f) {
  unsigned int u = __float_as_uint(f);
  unsigned int r = (u + 0x7FFFu + ((u >> 16) & 1u)) >> 16;   // round-to-nearest-even
  return (unsigned short)r;
}
__device__ __forceinline__ float sigf(float x) { return 1.0f / (1.0f + expf(-x)); }

// A-matrix 16x32 bf16 tile, contiguous K order:
// lanes 0-15 hold K=[0..15]; lanes 16-31 hold K=[16..31]
// (caller pre-offsets the row pointer by +16 elements for the upper half-wave)
__device__ __forceinline__ bf16x16 load_a16(const __bf16* p) {
  union { bf16x8 h[2]; bf16x16 v; } u;
  u.h[0] = *(const bf16x8*)(p);        // K block +0..7
  u.h[1] = *(const bf16x8*)(p + 8);    // K block +8..15
  return u.v;
}

// ---------- one-time prep kernels ----------
__global__ void cvt_wl_kernel(const float4* __restrict__ src, uint2* __restrict__ dst, int n4) {
  int i = blockIdx.x * 256 + threadIdx.x;
  if (i >= n4) return;
  float4 f = src[i];
  uint2 o;
  o.x = (unsigned)f2bf(f.x) | ((unsigned)f2bf(f.y) << 16);
  o.y = (unsigned)f2bf(f.z) | ((unsigned)f2bf(f.w) << 16);
  dst[i] = o;
}

__global__ void build_wg_kernel(const float* __restrict__ W_ih, const float* __restrict__ W_hh,
                                unsigned short* __restrict__ wg) {
  int idx = blockIdx.x * 256 + threadIdx.x;
  if (idx >= NG * KG) return;
  int n = idx / KG, k = idx % KG;
  float v = (k < Ed + Md) ? W_ih[(size_t)n * (Ed + Md) + k]
                          : W_hh[(size_t)n * Hd + (k - Ed - Md)];
  wg[idx] = f2bf(v);
}

// s_enc[b,k] = enc[b,k,:] . Wa[2H:2H+E]   (time-invariant part of attention scores)
__global__ void senc_kernel(const float* __restrict__ enc, const float* __restrict__ Wa,
                            float* __restrict__ s_enc) {
  int bk = blockIdx.x;            // b*Kd + k
  int tid = threadIdx.x;          // 64 threads
  __shared__ float red[64];
  float a = 0.f;
  for (int e = tid; e < Ed; e += 64) a += enc[(size_t)bk * Ed + e] * Wa[2 * Hd + e];
  red[tid] = a; __syncthreads();
  for (int s = 32; s > 0; s >>= 1) { if (tid < s) red[tid] += red[tid + s]; __syncthreads(); }
  if (tid == 0) s_enc[bk] = red[0];
}

__global__ void init_hc_kernel(const float* __restrict__ enc, float* __restrict__ h, float* __restrict__ c) {
  int idx = blockIdx.x * 256 + threadIdx.x;
  if (idx >= Bd * Hd) return;
  int b = idx / Hd, e = idx % Hd;
  float v = enc[((size_t)b * Kd + (Kd - 1)) * Ed + e];
  h[idx] = v; c[idx] = v;
}

// ---------- per-step kernels ----------
// Attention + context + embedding gather; packs bf16 operand vectors for both GEMMs.
__global__ void attn_kernel(const float* __restrict__ enc, const float* __restrict__ embt,
                            const float* __restrict__ Wa, const float* __restrict__ ba,
                            const int* __restrict__ caps, const int* __restrict__ tf_flag,
                            const int* __restrict__ next_tok, const float* __restrict__ s_enc,
                            const float* __restrict__ h, const float* __restrict__ c,
                            unsigned short* __restrict__ xh_bf, unsigned short* __restrict__ fin_bf,
                            int t) {
  int b = blockIdx.x, tid = threadIdx.x;
  __shared__ float red[256];
  __shared__ float attnw[Kd];
  __shared__ float s_stat, s_max, s_sum;

  // state part of the attention score: [h,c] . Wa[0:2H]
  float acc = 0.f;
  for (int i = tid; i < Hd; i += 256)
    acc += h[b * Hd + i] * Wa[i] + c[b * Hd + i] * Wa[Hd + i];
  red[tid] = acc; __syncthreads();
  for (int s = 128; s > 0; s >>= 1) { if (tid < s) red[tid] += red[tid + s]; __syncthreads(); }
  if (tid == 0) s_stat = red[0] + ba[0];
  __syncthreads();

  if (tid < Kd) attnw[tid] = tanhf(s_stat + s_enc[b * Kd + tid]);
  __syncthreads();
  if (tid == 0) { float mx = attnw[0]; for (int k = 1; k < Kd; ++k) mx = fmaxf(mx, attnw[k]); s_max = mx; }
  __syncthreads();
  if (tid < Kd) attnw[tid] = expf(attnw[tid] - s_max);
  __syncthreads();
  if (tid == 0) { float s = 0.f; for (int k = 0; k < Kd; ++k) s += attnw[k]; s_sum = s; }
  __syncthreads();
  if (tid < Kd) attnw[tid] = attnw[tid] / s_sum;
  __syncthreads();

  int tf  = tf_flag[0];
  int tok = (t == 0) ? caps[b * Td] : (tf ? caps[b * Td + (t - 1)] : next_tok[b]);

  // context[e] = sum_k attn[k] * enc[b,k,e]; pack into xh (=[ctx,emb,h]) and fin (=[h,c,ctx,emb])
  for (int e = tid; e < Ed; e += 256) {
    float cv = 0.f;
    #pragma unroll 4
    for (int k = 0; k < Kd; ++k) cv += attnw[k] * enc[((size_t)(b * Kd + k)) * Ed + e];
    unsigned short cb = f2bf(cv);
    xh_bf[b * KG + e]            = cb;
    fin_bf[b * KL + 2 * Hd + e]  = cb;
  }
  for (int i = tid; i < Md; i += 256) {
    unsigned short eb = f2bf(embt[(size_t)tok * Md + i]);
    xh_bf[b * KG + Ed + i]            = eb;
    fin_bf[b * KL + 2 * Hd + Ed + i]  = eb;
  }
  for (int i = tid; i < Hd; i += 256) {
    unsigned short hb = f2bf(h[b * Hd + i]);
    unsigned short cb = f2bf(c[b * Hd + i]);
    xh_bf[b * KG + Ed + Md + i] = hb;
    fin_bf[b * KL + i]          = hb;
    fin_bf[b * KL + Hd + i]     = cb;
  }
}

// gates = xh(32xKG) @ Wg^T(NGxKG) + b_ih + b_hh, via v_wmma_f32_16x16x32_bf16
__global__ void gates_wmma_kernel(const unsigned short* __restrict__ xh_bf,
                                  const unsigned short* __restrict__ wg_bf,
                                  const float* __restrict__ b_ih, const float* __restrict__ b_hh,
                                  float* __restrict__ gates) {
  const int lane   = threadIdx.x & 31;
  const int wave   = threadIdx.x >> 5;
  const int n_base = blockIdx.x * 128 + wave * 16;
  const int n  = lane & 15;
  const int hi = lane >> 4;
  const __bf16* A  = (const __bf16*)xh_bf;
  const __bf16* Bm = (const __bf16*)wg_bf;
  const __bf16* a0 = A + (size_t)(lane & 15) * KG + hi * 16;  // rows 0..15
  const __bf16* a1 = a0 + (size_t)16 * KG;                    // rows 16..31
  const __bf16* br = Bm + (size_t)(n_base + n) * KG + hi * 16;
  f32x8 acc0 = {}, acc1 = {};
  for (int kb = 0; kb < KG; kb += 32) {
    bf16x16 av0 = load_a16(a0 + kb);
    bf16x16 av1 = load_a16(a1 + kb);
    bf16x16 bv  = *(const bf16x16*)(br + kb);
    acc0 = __builtin_amdgcn_wmma_f32_16x16x32_bf16(false, av0, false, bv, (short)0, acc0, false, false);
    acc1 = __builtin_amdgcn_wmma_f32_16x16x32_bf16(false, av1, false, bv, (short)0, acc1, false, false);
  }
  const int gn = n_base + n;
  const float bias = b_ih[gn] + b_hh[gn];
  const int mb = hi * 8;
  #pragma unroll
  for (int j = 0; j < 8; ++j) {
    gates[(size_t)(mb + j) * NG + gn]      = acc0[j] + bias;
    gates[(size_t)(mb + j + 16) * NG + gn] = acc1[j] + bias;
  }
}

__global__ void lstm_kernel(const float* __restrict__ gates, float* __restrict__ h, float* __restrict__ c) {
  int idx = blockIdx.x * 256 + threadIdx.x;
  if (idx >= Bd * Hd) return;
  int b = idx / Hd, n = idx % Hd;
  const float* g = gates + (size_t)b * NG;
  float ig = g[n], fg = g[Hd + n], gg = g[2 * Hd + n], og = g[3 * Hd + n];
  float cn = sigf(fg) * c[idx] + sigf(ig) * tanhf(gg);
  c[idx] = cn;
  h[idx] = sigf(og) * tanhf(cn);
}

// logits = fin(32xKL) @ Wl_bf16^T(VdxKL) + bl, via v_wmma_f32_16x16x32_bf16
__global__ void logits_wmma_kernel(const unsigned short* __restrict__ fin_bf,
                                   const unsigned short* __restrict__ wl_bf,
                                   const float* __restrict__ bl, float* __restrict__ logits) {
  const int lane   = threadIdx.x & 31;
  const int wave   = threadIdx.x >> 5;
  const int v_base = blockIdx.x * 128 + wave * 16;
  const int n  = lane & 15;
  const int hi = lane >> 4;
  const __bf16* A  = (const __bf16*)fin_bf;
  const __bf16* Bm = (const __bf16*)wl_bf;
  const __bf16* a0 = A + (size_t)(lane & 15) * KL + hi * 16;
  const __bf16* a1 = a0 + (size_t)16 * KL;
  const __bf16* br = Bm + (size_t)(v_base + n) * KL + hi * 16;
  f32x8 acc0 = {}, acc1 = {};
  for (int kb = 0; kb < KL; kb += 32) {
    if (kb + 256 < KL) __builtin_prefetch(br + kb + 256, 0, 1);  // global_prefetch_b8
    bf16x16 av0 = load_a16(a0 + kb);
    bf16x16 av1 = load_a16(a1 + kb);
    bf16x16 bv  = *(const bf16x16*)(br + kb);
    acc0 = __builtin_amdgcn_wmma_f32_16x16x32_bf16(false, av0, false, bv, (short)0, acc0, false, false);
    acc1 = __builtin_amdgcn_wmma_f32_16x16x32_bf16(false, av1, false, bv, (short)0, acc1, false, false);
  }
  const int v = v_base + n;
  const float bias = bl[v];
  const int mb = hi * 8;
  #pragma unroll
  for (int j = 0; j < 8; ++j) {
    logits[(size_t)(mb + j) * Vd + v]      = acc0[j] + bias;
    logits[(size_t)(mb + j + 16) * Vd + v] = acc1[j] + bias;
  }
}

// per-row log_softmax over V=32000 + argmax (feeds next token when tf==0)
__global__ void logsoftmax_kernel(const float* __restrict__ logits, float* __restrict__ out,
                                  int* __restrict__ next_tok, int t) {
  int b = blockIdx.x, tid = threadIdx.x;
  __shared__ float rmax[256]; __shared__ int rarg[256]; __shared__ float rsum[256];
  float m = -3.4e38f; int mi = 0;
  for (int v = tid; v < Vd; v += 256) {
    float x = logits[(size_t)b * Vd + v];
    if (x > m) { m = x; mi = v; }
  }
  rmax[tid] = m; rarg[tid] = mi; __syncthreads();
  for (int s = 128; s > 0; s >>= 1) {
    if (tid < s) {
      if (rmax[tid + s] > rmax[tid] || (rmax[tid + s] == rmax[tid] && rarg[tid + s] < rarg[tid])) {
        rmax[tid] = rmax[tid + s]; rarg[tid] = rarg[tid + s];
      }
    }
    __syncthreads();
  }
  float mx = rmax[0];
  float s = 0.f;
  for (int v = tid; v < Vd; v += 256) s += expf(logits[(size_t)b * Vd + v] - mx);
  rsum[tid] = s; __syncthreads();
  for (int st = 128; st > 0; st >>= 1) { if (tid < st) rsum[tid] += rsum[tid + st]; __syncthreads(); }
  float lse = mx + logf(rsum[0]);
  float* orow = out + ((size_t)b * Td + t) * Vd;
  for (int v = tid; v < Vd; v += 256) orow[v] = logits[(size_t)b * Vd + v] - lse;
  if (tid == 0) next_tok[b] = rarg[0];
}

// ---------- host ----------
extern "C" void kernel_launch(void* const* d_in, const int* in_sizes, int n_in,
                              void* d_out, int out_size, void* d_ws, size_t ws_size,
                              hipStream_t stream) {
  (void)in_sizes; (void)n_in; (void)out_size; (void)ws_size;
  const float* enc  = (const float*)d_in[0];
  const float* embt = (const float*)d_in[1];
  const float* Wa   = (const float*)d_in[2];
  const float* ba   = (const float*)d_in[3];
  const float* W_ih = (const float*)d_in[4];
  const float* W_hh = (const float*)d_in[5];
  const float* b_ih = (const float*)d_in[6];
  const float* b_hh = (const float*)d_in[7];
  const float* Wl   = (const float*)d_in[8];
  const float* bl   = (const float*)d_in[9];
  const int*   caps = (const int*)d_in[10];
  const int*   tf   = (const int*)d_in[11];
  float* out = (float*)d_out;

  // workspace carve-up (~175 MB total); 256B aligned chunks
  char* p = (char*)d_ws;
  auto take = [&](size_t bytes) { char* r = p; p += (bytes + 255) & ~(size_t)255; return r; };
  unsigned short* wl_bf   = (unsigned short*)take((size_t)Vd * KL * 2); // 164 MB -> fits L2
  unsigned short* wg_bf   = (unsigned short*)take((size_t)NG * KG * 2);
  float*          s_enc   = (float*)take((size_t)Bd * Kd * 4);
  float*          hbuf    = (float*)take((size_t)Bd * Hd * 4);
  float*          cbuf    = (float*)take((size_t)Bd * Hd * 4);
  unsigned short* xh_bf   = (unsigned short*)take((size_t)Bd * KG * 2);
  unsigned short* fin_bf  = (unsigned short*)take((size_t)Bd * KL * 2);
  float*          gates   = (float*)take((size_t)Bd * NG * 4);
  float*          logits  = (float*)take((size_t)Bd * Vd * 4);
  int*            nexttok = (int*)take((size_t)Bd * 4);

  // one-time prep
  int n4 = Vd * KL / 4;
  cvt_wl_kernel<<<(n4 + 255) / 256, 256, 0, stream>>>((const float4*)Wl, (uint2*)wl_bf, n4);
  build_wg_kernel<<<(NG * KG + 255) / 256, 256, 0, stream>>>(W_ih, W_hh, wg_bf);
  senc_kernel<<<Bd * Kd, 64, 0, stream>>>(enc, Wa, s_enc);
  init_hc_kernel<<<(Bd * Hd + 255) / 256, 256, 0, stream>>>(enc, hbuf, cbuf);

  // sequential decode steps
  for (int t = 0; t < Td; ++t) {
    attn_kernel<<<Bd, 256, 0, stream>>>(enc, embt, Wa, ba, caps, tf, nexttok,
                                        s_enc, hbuf, cbuf, xh_bf, fin_bf, t);
    gates_wmma_kernel<<<NG / 128, 256, 0, stream>>>(xh_bf, wg_bf, b_ih, b_hh, gates);
    lstm_kernel<<<(Bd * Hd + 255) / 256, 256, 0, stream>>>(gates, hbuf, cbuf);
    logits_wmma_kernel<<<Vd / 128, 256, 0, stream>>>(fin_bf, wl_bf, bl, logits);
    logsoftmax_kernel<<<Bd, 256, 0, stream>>>(logits, out, nexttok, t);
  }
}
